// MambaStack_28123445854417
// MI455X (gfx1250) — matrix operand order repaired and checked by reading, not compiled, not run on hardware
//
#include <hip/hip_runtime.h>
#include <hip/hip_bf16.h>

// ---------------------------------------------------------------------------
// Mamba stack for MI455X (gfx1250, wave32).
// GEMMs: bf16 WMMA (v_wmma_f32_16x16x32_bf16), register-blocked 2x4 tiles per
// wave (8 WMMAs per K-step). Strides/K are template constants so global loads
// use SGPR-base + 32-bit VGPR offset + immediate (no 64-bit per-lane pointer
// chains -> no spills). fp32->bf16 conversion in registers, fp32 accumulate.
// Scan: one wave per (b, d_inner) channel, 64 states over 32 lanes (2/lane),
// sequential over L with shfl-xor reduction for y.
// ---------------------------------------------------------------------------

typedef __attribute__((ext_vector_type(16))) __bf16 v16bf;
typedef __attribute__((ext_vector_type(8)))  float  v8f;

#define BB        2
#define LL        2048
#define DMODEL    1024
#define NLAYERS   4
#define DINNER    2048
#define DSTATE    64
#define DCONV     4
#define DTRANK    64
#define MTOK      (BB*LL)             // 4096 token rows
#define XZLD      (2*DINNER)          // 4096
#define DBLLD     (DTRANK + 2*DSTATE) // 192

__device__ __forceinline__ float sigmoidf_(float v) {
    return 1.0f / (1.0f + __expf(-v));
}
__device__ __forceinline__ float n2n(float v) {
    if (v != v) return 0.0f;                       // NaN -> 0
    if (v >  3.402823466e38f) return  3.402823466e38f;
    if (v < -3.402823466e38f) return -3.402823466e38f;
    return v;
}
__device__ __forceinline__ float softplusf_(float v) {
    return (v > 20.0f) ? v : __logf(1.0f + __expf(v));
}

// 16 contiguous fp32 at (base + off elements) -> bf16 fragment (B rows)
__device__ __forceinline__ v16bf load_frag16(const float* __restrict__ p, int off) {
    const float4 x0 = *(const float4*)(p + off);
    const float4 x1 = *(const float4*)(p + off + 4);
    const float4 x2 = *(const float4*)(p + off + 8);
    const float4 x3 = *(const float4*)(p + off + 12);
    v16bf f;
    f[0]=(__bf16)x0.x; f[1]=(__bf16)x0.y; f[2]=(__bf16)x0.z; f[3]=(__bf16)x0.w;
    f[4]=(__bf16)x1.x; f[5]=(__bf16)x1.y; f[6]=(__bf16)x1.z; f[7]=(__bf16)x1.w;
    f[8]=(__bf16)x2.x; f[9]=(__bf16)x2.y; f[10]=(__bf16)x2.z; f[11]=(__bf16)x2.w;
    f[12]=(__bf16)x3.x; f[13]=(__bf16)x3.y; f[14]=(__bf16)x3.z; f[15]=(__bf16)x3.w;
    return f;
}
// A fragment: 8 fp32 at (off), 8 fp32 at (off+16)  (ISA 7.12.2 A layout)
__device__ __forceinline__ v16bf load_fragA(const float* __restrict__ p, int off) {
    const float4 x0 = *(const float4*)(p + off);
    const float4 x1 = *(const float4*)(p + off + 4);
    const float4 x2 = *(const float4*)(p + off + 16);
    const float4 x3 = *(const float4*)(p + off + 20);
    v16bf f;
    f[0]=(__bf16)x0.x; f[1]=(__bf16)x0.y; f[2]=(__bf16)x0.z; f[3]=(__bf16)x0.w;
    f[4]=(__bf16)x1.x; f[5]=(__bf16)x1.y; f[6]=(__bf16)x1.z; f[7]=(__bf16)x1.w;
    f[8]=(__bf16)x2.x; f[9]=(__bf16)x2.y; f[10]=(__bf16)x2.z; f[11]=(__bf16)x2.w;
    f[12]=(__bf16)x3.x; f[13]=(__bf16)x3.y; f[14]=(__bf16)x3.z; f[15]=(__bf16)x3.w;
    return f;
}

// ---------------------------------------------------------------------------
// Blocked bf16-WMMA GEMM:  C[M,N] = A[M,K] * W[N,K]^T   (+ epilogue)
// Each wave: 32(M) x 64(N) output region = 2x4 tiles of 16x16, 8 WMMAs/K-step.
// Block (32,4): 4 waves stacked on M -> block region 128(M) x 64(N).
// Grid: (N/64, M/128).  Requires K%32==0, N%64==0, M%128==0 (all call sites).
// LDA/LDW/LDC/K compile-time: row displacements fold into load immediates,
// only two 32-bit per-lane offsets stay live in the K loop.
// EPI 0: plain store
// EPI 1: softplus(acc + bias[n])                  (dt projection)
// EPI 2: C[m,n] = n2n(resid + n2n(acc)*keep[m])   (out_proj + residual + mask)
// ---------------------------------------------------------------------------
template <int EPI, int LDA, int LDW, int LDC, int K>
__global__ void gemm_wmma_bf16(const float* __restrict__ A,
                               const float* __restrict__ W,
                               float* C,
                               const float* __restrict__ bias,
                               const float* resid,
                               const unsigned char* __restrict__ kpm)
{
    const int lane  = threadIdx.x;                         // 0..31
    const int mBase = blockIdx.y * 128 + threadIdx.y * 32; // wave's 32-row strip
    const int nBase = blockIdx.x * 64;                     // wave's 64-col strip

    const int koffA = (lane & 16) ? 8 : 0;    // lanes16-31: K=8..15 / 24..31
    const int koffB = (lane & 16) ? 8 : 0;    // lanes16-31: K=8..15 / 24..31

    int aOff = (mBase + (lane & 15)) * LDA + koffA;   // 32-bit element offsets
    int wOff = (nBase + (lane & 15)) * LDW + koffB;

    v8f acc[2][4] = {};

    for (int k0 = 0; k0 < K; k0 += 32, aOff += 32, wOff += 32) {
        const v16bf fa0 = load_fragA(A, aOff);
        const v16bf fa1 = load_fragA(A, aOff + 16 * LDA);
        const v16bf fb0 = load_fragA(W, wOff);
        const v16bf fb1 = load_fragA(W, wOff + 16 * LDW);
        const v16bf fb2 = load_fragA(W, wOff + 32 * LDW);
        const v16bf fb3 = load_fragA(W, wOff + 48 * LDW);

        acc[0][0] = __builtin_amdgcn_wmma_f32_16x16x32_bf16(false, fa0, false, fb0, (short)0, acc[0][0], false, false);
        acc[0][1] = __builtin_amdgcn_wmma_f32_16x16x32_bf16(false, fa0, false, fb1, (short)0, acc[0][1], false, false);
        acc[0][2] = __builtin_amdgcn_wmma_f32_16x16x32_bf16(false, fa0, false, fb2, (short)0, acc[0][2], false, false);
        acc[0][3] = __builtin_amdgcn_wmma_f32_16x16x32_bf16(false, fa0, false, fb3, (short)0, acc[0][3], false, false);
        acc[1][0] = __builtin_amdgcn_wmma_f32_16x16x32_bf16(false, fa1, false, fb0, (short)0, acc[1][0], false, false);
        acc[1][1] = __builtin_amdgcn_wmma_f32_16x16x32_bf16(false, fa1, false, fb1, (short)0, acc[1][1], false, false);
        acc[1][2] = __builtin_amdgcn_wmma_f32_16x16x32_bf16(false, fa1, false, fb2, (short)0, acc[1][2], false, false);
        acc[1][3] = __builtin_amdgcn_wmma_f32_16x16x32_bf16(false, fa1, false, fb3, (short)0, acc[1][3], false, false);
    }

    // D layout: VGPR v -> M = base + v (+8 for lanes 16-31), N = base + (lane&15)
    const int mAdd = (lane & 16) ? 8 : 0;
#pragma unroll
    for (int i = 0; i < 2; ++i) {
#pragma unroll
        for (int j = 0; j < 4; ++j) {
            const int n = nBase + j * 16 + (lane & 15);
#pragma unroll
            for (int v = 0; v < 8; ++v) {
                const int m = mBase + i * 16 + v + mAdd;
                float val = acc[i][j][v];
                if (EPI == 1) {
                    val = softplusf_(val + bias[n]);
                } else if (EPI == 2) {
                    const float keep = kpm[m] ? 0.0f : 1.0f;
                    val = n2n(resid[m * LDC + n] + n2n(val) * keep);
                }
                C[m * LDC + n] = val;
            }
        }
    }
}

// ---------------------------------------------------------------------------
// x0 = x * keep
// ---------------------------------------------------------------------------
__global__ void init_mask_kernel(const float* __restrict__ x,
                                 const unsigned char* __restrict__ kpm,
                                 float* __restrict__ bufX)
{
    const size_t i = (size_t)blockIdx.x * blockDim.x + threadIdx.x;
    if (i >= (size_t)MTOK * DMODEL) return;
    const size_t row = i / DMODEL;
    bufX[i] = kpm[row] ? 0.0f : x[i];
}

// ---------------------------------------------------------------------------
// Depthwise causal conv (4 taps) over L + bias + SiLU.  xin = xz[..., :DINNER]
// ---------------------------------------------------------------------------
__global__ void conv_silu_kernel(const float* __restrict__ xz,
                                 const float* __restrict__ cw,
                                 const float* __restrict__ cb,
                                 float* __restrict__ xc)
{
    const size_t i = (size_t)blockIdx.x * blockDim.x + threadIdx.x;
    if (i >= (size_t)MTOK * DINNER) return;
    const int d = (int)(i % DINNER);
    const int l = (int)((i / DINNER) % LL);
    const int b = (int)(i / ((size_t)DINNER * LL));

    const float4 w = *(const float4*)(cw + d * 4);
    float acc = cb[d];
    const size_t rowBase = ((size_t)b * LL) * XZLD + d;
    if (l >= 3) acc += w.x * xz[rowBase + (size_t)(l - 3) * XZLD];
    if (l >= 2) acc += w.y * xz[rowBase + (size_t)(l - 2) * XZLD];
    if (l >= 1) acc += w.z * xz[rowBase + (size_t)(l - 1) * XZLD];
    acc += w.w * xz[rowBase + (size_t)l * XZLD];
    xc[i] = acc * sigmoidf_(acc);
}

// ---------------------------------------------------------------------------
// Selective scan. One wave per (b, d) channel; 64 states over 32 lanes.
//   h_n(t) = h_n(t-1)*exp(dt*A_n) + dt*x*B_n(t);  y(t) = sum_n h_n*C_n(t)
// Epilogue: y = (y + x*Dp) * silu(z).  yOut may alias dtIn (safe: same wave
// reads dt[t] before writing y[t]).
// ---------------------------------------------------------------------------
__global__ void scan_kernel(const float* __restrict__ xc, const float* dtIn,
                            const float* __restrict__ dbl,
                            const float* __restrict__ xz,
                            const float* __restrict__ A_log,
                            const float* __restrict__ Dpar,
                            float* yOut)
{
    const int ch = blockIdx.x * blockDim.y + threadIdx.y;   // 0..B*DINNER-1
    const int b  = ch >> 11;            // / DINNER
    const int d  = ch & (DINNER - 1);
    const int lane = threadIdx.x;
    const int n0 = lane, n1 = lane + 32;

    const float Aa = -__expf(A_log[(size_t)d * DSTATE + n0]);
    const float Ab = -__expf(A_log[(size_t)d * DSTATE + n1]);
    const float Dp = Dpar[d];

    float h0 = 0.0f, h1 = 0.0f;
    for (int t = 0; t < LL; ++t) {
        const size_t tok = (size_t)b * LL + t;
        const float xv  = xc  [tok * DINNER + d];
        const float dtv = dtIn[tok * DINNER + d];
        const float Bv0 = dbl[tok * DBLLD + DTRANK + n0];
        const float Bv1 = dbl[tok * DBLLD + DTRANK + n1];
        const float Cv0 = dbl[tok * DBLLD + DTRANK + DSTATE + n0];
        const float Cv1 = dbl[tok * DBLLD + DTRANK + DSTATE + n1];

        const float du = dtv * xv;
        h0 = h0 * __expf(dtv * Aa) + du * Bv0;
        h1 = h1 * __expf(dtv * Ab) + du * Bv1;
        float acc = h0 * Cv0 + h1 * Cv1;
#pragma unroll
        for (int off = 16; off > 0; off >>= 1)
            acc += __shfl_xor(acc, off, 32);

        if (lane == 0) {
            const float z = xz[tok * XZLD + DINNER + d];
            yOut[tok * DINNER + d] = (acc + xv * Dp) * (z * sigmoidf_(z));
        }
    }
}

// ---------------------------------------------------------------------------
// Final LayerNorm over DMODEL + mask + nan_to_num.
// ---------------------------------------------------------------------------
__global__ void ln_kernel(const float* __restrict__ xin,
                          const float* __restrict__ gamma,
                          const float* __restrict__ beta,
                          const unsigned char* __restrict__ kpm,
                          float* __restrict__ out)
{
    const int row = blockIdx.x;
    const int tid = threadIdx.x;
    const float* xr = xin + (size_t)row * DMODEL;
    __shared__ float s1[256], s2[256];
    float sum = 0.0f, sq = 0.0f;
    for (int c = tid; c < DMODEL; c += 256) {
        const float v = xr[c];
        sum += v; sq += v * v;
    }
    s1[tid] = sum; s2[tid] = sq;
    __syncthreads();
    for (int st = 128; st > 0; st >>= 1) {
        if (tid < st) { s1[tid] += s1[tid + st]; s2[tid] += s2[tid + st]; }
        __syncthreads();
    }
    const float mu  = s1[0] * (1.0f / DMODEL);
    const float var = s2[0] * (1.0f / DMODEL) - mu * mu;
    const float inv = rsqrtf(var + 1e-5f);
    const float keep = kpm[row] ? 0.0f : 1.0f;
    for (int c = tid; c < DMODEL; c += 256) {
        const float v = (xr[c] - mu) * inv * gamma[c] + beta[c];
        out[(size_t)row * DMODEL + c] = n2n(n2n(v) * keep);
    }
}

// ---------------------------------------------------------------------------
extern "C" void kernel_launch(void* const* d_in, const int* in_sizes, int n_in,
                              void* d_out, int out_size, void* d_ws, size_t ws_size,
                              hipStream_t stream)
{
    const float*         x      = (const float*)d_in[0];
    const unsigned char* kpm    = (const unsigned char*)d_in[1]; // bool mask
    const float*         in_w   = (const float*)d_in[2];
    const float*         conv_w = (const float*)d_in[3];
    const float*         conv_b = (const float*)d_in[4];
    const float*         xp_w   = (const float*)d_in[5];
    const float*         dt_w   = (const float*)d_in[6];
    const float*         dt_b   = (const float*)d_in[7];
    const float*         A_log  = (const float*)d_in[8];
    const float*         Dpar   = (const float*)d_in[9];
    const float*         out_w  = (const float*)d_in[10];
    const float*         gamma  = (const float*)d_in[11];
    const float*         beta   = (const float*)d_in[12];

    float* ws     = (float*)d_ws;
    float* bufX   = ws;                                   //  4M f32 (running x)
    float* bufXZ  = bufX  + (size_t)MTOK * DMODEL;        // 16M f32
    float* bufXC  = bufXZ + (size_t)MTOK * XZLD;          //  8M f32
    float* bufDBL = bufXC + (size_t)MTOK * DINNER;        // 768K f32
    float* bufDT  = bufDBL + (size_t)MTOK * DBLLD;        //  8M f32 (dt, then y)

    const dim3 gblk(32, 4);     // 4 waves per block, 32x64 output region each

    // x0 = x * keep
    {
        const int tot = MTOK * DMODEL;
        init_mask_kernel<<<(tot + 255) / 256, 256, 0, stream>>>(x, kpm, bufX);
    }

    for (int layer = 0; layer < NLAYERS; ++layer) {
        const float* Wi  = in_w  + (size_t)layer * XZLD * DMODEL;
        const float* Wx  = xp_w  + (size_t)layer * DBLLD * DINNER;
        const float* Wdt = dt_w  + (size_t)layer * DINNER * DTRANK;
        const float* Wo  = out_w + (size_t)layer * DMODEL * DINNER;

        // xz = x @ Wi^T           [4096 x 4096], K=1024
        gemm_wmma_bf16<0, DMODEL, DMODEL, XZLD, DMODEL>
            <<<dim3(XZLD / 64, MTOK / 128), gblk, 0, stream>>>(
                bufX, Wi, bufXZ, nullptr, nullptr, nullptr);

        // depthwise conv + SiLU -> xc
        {
            const int tot = MTOK * DINNER;
            conv_silu_kernel<<<(tot + 255) / 256, 256, 0, stream>>>(
                bufXZ, conv_w + (size_t)layer * DINNER * DCONV,
                conv_b + (size_t)layer * DINNER, bufXC);
        }

        // dbl = xc @ Wx^T         [4096 x 192], K=2048
        gemm_wmma_bf16<0, DINNER, DINNER, DBLLD, DINNER>
            <<<dim3(DBLLD / 64, MTOK / 128), gblk, 0, stream>>>(
                bufXC, Wx, bufDBL, nullptr, nullptr, nullptr);

        // dt = softplus(dtr @ Wdt^T + b)   [4096 x 2048], K=64 (lda=192)
        gemm_wmma_bf16<1, DBLLD, DTRANK, DINNER, DTRANK>
            <<<dim3(DINNER / 64, MTOK / 128), gblk, 0, stream>>>(
                bufDBL, Wdt, bufDT, dt_b + (size_t)layer * DINNER,
                nullptr, nullptr);

        // selective scan -> y (in place over dt buffer)
        scan_kernel<<<dim3((BB * DINNER) / 4), dim3(32, 4), 0, stream>>>(
            bufXC, bufDT, bufDBL, bufXZ,
            A_log + (size_t)layer * DINNER * DSTATE,
            Dpar  + (size_t)layer * DINNER, bufDT);

        // x = n2n(x + n2n(y @ Wo^T) * keep)   [4096 x 1024], K=2048
        gemm_wmma_bf16<2, DINNER, DINNER, DMODEL, DINNER>
            <<<dim3(DMODEL / 64, MTOK / 128), gblk, 0, stream>>>(
                bufDT, Wo, bufX, nullptr, bufX, kpm);
    }

    // LayerNorm + mask -> out
    ln_kernel<<<MTOK, 256, 0, stream>>>(bufX, gamma, beta, kpm, (float*)d_out);
}
